// ConvColumn_17214228922889
// MI455X (gfx1250) — matrix operand order repaired and checked by reading, not compiled, not run on hardware
//
#include <hip/hip_runtime.h>
#include <cstdint>

// ---------------- problem geometry ----------------
#define B_      2
#define IN_C_   8
#define OUT_C_  32
#define X_      96
#define T_IN    96
#define NXY     47                 // (96-3)/2 + 1
#define NSITE   (NXY*NXY)          // 2209
#define K_T_    48
#define PAD_    32
#define T_OUT   113                // 96 + 2*32 - 48 + 1
#define T_PAD   128
#define NROW    (IN_C_*3*3)        // 72 (i,kx,ky) rows
#define TROW    176                // staged time axis: t-index 0..175 (covers pad-wave reads)
#define KTOT    (NROW*K_T_)        // 3456
#define KCHUNKS (KTOT/32)          // 108
#define NCOPY   4                  // shifted LDS copies -> 8B-aligned B fragments

typedef __attribute__((ext_vector_type(16))) _Float16 v16h;
typedef __attribute__((ext_vector_type(8)))  float    v8f;

#define POT_ELEMS ((size_t)OUT_C_*B_*NSITE*T_PAD)   // 18.1M floats = 72.4 MB (L2-resident)
#define POT_BYTES (POT_ELEMS*4)

// ---------------------------------------------------------------------------
// Kernel 1: expand temporal weight kernel straight into WMMA A-fragment layout.
// A is M=32 x K=3456 (two 16-row M tiles). Fragment layout (contiguous k,
// same order as B): lane<16 : K=j
//                lane>=16: K=j+16
// k = row*48 + kt, row = i*9 + kx*3 + ky; tap kt multiplies x[t = ot+kt-32]
// with kernel value v(47-kt) (python kern is time-reversed; XLA conv is
// cross-correlation).
// ---------------------------------------------------------------------------
__global__ __launch_bounds__(256)
void expand_weight_frags(const float* __restrict__ weight, _Float16* __restrict__ Afrag) {
  int tid = blockIdx.x*256 + threadIdx.x;        // one thread = one (kc, mt, lane)
  if (tid >= KCHUNKS*2*32) return;
  int lane = tid & 31;
  int mt   = (tid >> 5) & 1;
  int kc   = tid >> 6;
  int m = lane & 15;
  int o = mt*16 + m;
  bool hi = (lane & 16) != 0;
  v16h vv;
  #pragma unroll
  for (int j = 0; j < 16; ++j) {
    int kl = hi ? (j + 16)
                :  j;
    int k   = kc*32 + kl;
    int row = k / K_T_;
    int kt  = k - row*K_T_;
    float w = weight[o*NROW + row];              // weight[o][i][kx][ky]
    float t = (float)(K_T_ - 1 - kt);            // time-reversed tap
    float t_spike = t * (1.0f/16.0f);
    float t_leak  = w - (t - w*16.0f) * (1.0f/32.0f);
    float v = fmaxf(0.0f, fminf(t_spike, t_leak));
    vv[j] = (_Float16)v;
  }
  *(v16h*)(Afrag + (size_t)tid*16) = vv;         // tid == (kc*2+mt)*32+lane
}

// ---------------------------------------------------------------------------
// Kernel 2: implicit-GEMM conv via V_WMMA_F32_16X16X32_F16.
// One workgroup (8 waves) per (b, spatial site). Input slab (72 rows x 96 t)
// staged to LDS as f16, zero-padded, in NCOPY=4 copies shifted by 0..3
// halfwords: copy p holds S[p][i] = T[i+p] where T is the logical padded time
// axis. A lane's B fragment needs T[base .. base+15] with base = n (mod 4),
// so it reads S[n&3][base-(n&3) ..] at an 8B-aligned offset -> 4x ds_load_b64
// (full-rate LDS) instead of 8x ds_load_b32.
// B layout (dense 16-bit K=32, mirrors ISA sparse-B pattern):
//   lane L: n = L&15;  half j -> k = (L<16 ? j : 16+j).
// k-runs of 16 start at kt0 in {0,16,32} so they never cross a 48 boundary.
// ---------------------------------------------------------------------------
__global__ __launch_bounds__(256)
void snn_conv_wmma(const float* __restrict__ x, const _Float16* __restrict__ Afrag,
                   float* __restrict__ pot) {
  __shared__ _Float16 S[NCOPY][NROW*TROW];       // 4 * 25,344 B = 101,376 B
  int site = blockIdx.x;
  int b  = site / NSITE;
  int s2 = site - b*NSITE;
  int ox = s2 / NXY;
  int oy = s2 - ox*NXY;

  // stage input slab (f32 -> f16) into all 4 shifted copies
  for (int e = threadIdx.x; e < NROW*T_IN; e += 256) {
    int row = e / T_IN; int t = e - row*T_IN;
    int i  = row / 9;  int r9 = row - i*9;
    int kx = r9 / 3;   int ky = r9 - kx*3;
    int xx = 2*ox + kx, yy = 2*oy + ky;
    float v = x[((((size_t)b*IN_C_ + i)*X_ + xx)*X_ + yy)*T_IN + t];
    _Float16 h = (_Float16)v;
    #pragma unroll
    for (int p = 0; p < NCOPY; ++p)
      S[p][row*TROW + (PAD_ - p) + t] = h;       // copy p data at [32-p, 128-p)
  }
  // zero pads: copy p zeros at [0, 32-p) and [128-p, 176)  (80 per row)
  for (int e = threadIdx.x; e < NROW*80; e += 256) {
    int row = e / 80; int q = e - row*80;
    #pragma unroll
    for (int p = 0; p < NCOPY; ++p) {
      int i = (q < PAD_ - p) ? q : (96 + q);     // maps q onto both zero ranges
      S[p][row*TROW + i] = (_Float16)0.0f;
    }
  }
  __syncthreads();

  int lane = threadIdx.x & 31;
  int wave = threadIdx.x >> 5;
  int t0 = wave * 16;                            // this wave's 16 output timesteps
  int n  = lane & 15;
  int par = n & (NCOPY - 1);

  v8f acc0 = {};
  v8f acc1 = {};
  const v16h* ap = (const v16h*)Afrag;
  int row = 0;
  int kt0 = (lane & 16) ? 16 : 0;
  const _Float16* Sp = &S[par][0] - par;         // offsets from here are 0 mod 4 halves
  #pragma unroll 3
  for (int kc = 0; kc < KCHUNKS; ++kc) {
    v16h a0 = ap[lane];                          // 32B contiguous per lane
    v16h a1 = ap[32 + lane];
    ap += 64;
    // fragment = T[base .. base+15], base = row*176 + t0 + kt0 + n  (== n mod 4)
    const uint64_t* p64 = (const uint64_t*)(Sp + row*TROW + t0 + kt0 + n);
    union { uint64_t u[4]; v16h v; } bb;
    #pragma unroll
    for (int q = 0; q < 4; ++q) bb.u[q] = p64[q];   // 4x ds_load_b64, 8B aligned
    acc0 = __builtin_amdgcn_wmma_f32_16x16x32_f16(false, a0, false, bb.v,
                                                  (short)0, acc0, false, false);
    acc1 = __builtin_amdgcn_wmma_f32_16x16x32_f16(false, a1, false, bb.v,
                                                  (short)0, acc1, false, false);
    kt0 += 32;
    if (kt0 >= K_T_) { kt0 -= K_T_; ++row; }     // kt0 cycles {0,32,16} / {16,0,32}
  }

  // D layout: elem r -> M = r + (lane>=16 ? 8:0), N = lane&15
  int tOut  = t0 + n;                            // 0..127 (t>=113 is pad, ignored)
  int cbase = (lane & 16) ? 8 : 0;
  #pragma unroll
  for (int r = 0; r < 8; ++r) {
    int c0 = cbase + r;
    int c1 = 16 + cbase + r;
    pot[((size_t)(c0*B_ + b)*NSITE + s2)*T_PAD + tOut] = acc0[r];
    pot[((size_t)(c1*B_ + b)*NSITE + s2)*T_PAD + tOut] = acc1[r];
  }
}

// ---------------------------------------------------------------------------
// Kernel 3: zero the (very sparse) output.
// ---------------------------------------------------------------------------
__global__ void zero_out_kernel(float* __restrict__ out, int nElem) {
  int i = blockIdx.x*256 + threadIdx.x;
  if (i < nElem) out[i] = 0.0f;
}

// ---------------------------------------------------------------------------
// Kernel 4: winner-takes-all scan. Batches are independent -> one workgroup
// per b. 32 waves = 32 channels; lanes stride over the 2209 sites. dep[] is
// shared across channels per site (reference sums winners over C). Update
// reduces to: spiked(site) ? 47 : max(dep-1, 0).
// ---------------------------------------------------------------------------
__global__ __launch_bounds__(1024)
void wta_scan(const float* __restrict__ pot, float* __restrict__ out) {
  __shared__ int   dep[NSITE];
  __shared__ float winV[OUT_C_];
  __shared__ int   winN[OUT_C_];
  int b = blockIdx.x;
  int c = threadIdx.x >> 5;
  int lane = threadIdx.x & 31;
  for (int nn = threadIdx.x; nn < NSITE; nn += 1024) dep[nn] = 0;
  __syncthreads();
  const float THETA = 3.6000001f;                // 0.05 * 3*3*8
  const float* pc = pot + (size_t)(c*B_ + b)*NSITE*T_PAD;
  for (int t = 0; t < T_OUT; ++t) {
    // masked argmax over sites for this channel (first-index tie-break)
    float best = -1.0f; int bestN = 0;
    for (int nn = lane; nn < NSITE; nn += 32) {
      float p = pc[(size_t)nn*T_PAD + t];
      p = (dep[nn] == 0) ? p : 0.0f;
      if (p > best) { best = p; bestN = nn; }
    }
    #pragma unroll
    for (int off = 16; off > 0; off >>= 1) {
      float ov = __shfl_down(best, off);
      int   on = __shfl_down(bestN, off);
      if (ov > best || (ov == best && on < bestN)) { best = ov; bestN = on; }
    }
    if (lane == 0) { winV[c] = best; winN[c] = bestN; }
    __syncthreads();
    // decrement all depressions (reads of old dep are done)
    for (int nn = threadIdx.x; nn < NSITE; nn += 1024) {
      int d = dep[nn] - 1;
      dep[nn] = d > 0 ? d : 0;
    }
    __syncthreads();
    // winners: emit spike, re-arm depression (multiple waves writing 47 is benign)
    if (lane == 0 && winV[c] > THETA) {
      dep[winN[c]] = K_T_ - 1;                   // 47
      out[((size_t)(b*OUT_C_ + c)*NSITE + winN[c])*T_OUT + t] = 1.0f;
    }
    __syncthreads();
  }
}

// ---------------------------------------------------------------------------
extern "C" void kernel_launch(void* const* d_in, const int* in_sizes, int n_in,
                              void* d_out, int out_size, void* d_ws, size_t ws_size,
                              hipStream_t stream) {
  (void)in_sizes; (void)n_in; (void)ws_size;
  const float* x = (const float*)d_in[0];        // input_spikes (2,8,96,96,96) f32
  const float* w = (const float*)d_in[1];        // weight (32,8,3,3) f32
  float* out = (float*)d_out;                    // (2,32,47,47,113) f32
  float*    pot   = (float*)d_ws;                             // 72.4 MB
  _Float16* Afrag = (_Float16*)((char*)d_ws + POT_BYTES);     // 216 KB

  expand_weight_frags<<<(KCHUNKS*64 + 255)/256, 256, 0, stream>>>(w, Afrag);
  zero_out_kernel<<<(out_size + 255)/256, 256, 0, stream>>>(out, out_size);
  snn_conv_wmma<<<B_*NSITE, 256, 0, stream>>>(x, Afrag, pot);
  wta_scan<<<B_, 1024, 0, stream>>>(pot, out);
}
